// TATT_1_73770358276489
// MI455X (gfx1250) — matrix operand order repaired and checked by reading, not compiled, not run on hardware
//
#include <hip/hip_runtime.h>
#include <hip/hip_bf16.h>

typedef __attribute__((ext_vector_type(16))) _Float16 v16h;
typedef __attribute__((ext_vector_type(8)))  _Float16 v8h;
typedef __attribute__((ext_vector_type(8)))  float    v8f;
typedef __attribute__((ext_vector_type(4)))  float    v4f;

#define Bb 32
#define Cc 32
#define Nn 512
#define Tt 480

// ---- fragment loaders (contiguous per-lane chunks, ISA §7.12.2 layouts) ----
// A (16-bit, 16x32): lane L: M=L%16, element e -> K = e + (e>=8?8:0) + 8*(L/16)
//   => two contiguous 8-half chunks at k0+8*half and k0+16+8*half
__device__ __forceinline__ v16h loadA16(const _Float16* rowp, int k0, int half) {
  v8h lo = *(const v8h*)(rowp + k0 + 8 * half);
  v8h hi = *(const v8h*)(rowp + k0 + 16 + 8 * half);
  return __builtin_shufflevector(lo, hi, 0,1,2,3,4,5,6,7,8,9,10,11,12,13,14,15);
}
// B (16-bit, 32x16): lane L: N=L%16, element e -> K = e + (e>=8?8:0) + 8*(L/16)
//   => two contiguous 8-half chunks at k0+8*half and k0+16+8*half
__device__ __forceinline__ v16h loadB16(const _Float16* colp, int k0, int half) {
  v8h lo = *(const v8h*)(colp + k0 + 8 * half);
  v8h hi = *(const v8h*)(colp + k0 + 16 + 8 * half);
  return __builtin_shufflevector(lo, hi, 0,1,2,3,4,5,6,7,8,9,10,11,12,13,14,15);
}

// ---------------------------------------------------------------------------
// Kernel 0a: wT[c][n] = (f16) w[n][c]   (one-time 32KB transpose)
// ---------------------------------------------------------------------------
__global__ void k0_wT(const float* __restrict__ w, _Float16* __restrict__ wT) {
  const int i = blockIdx.x * 256 + threadIdx.x;   // 0..16383
  const int n = i >> 5, c = i & 31;
  wT[(size_t)c * Nn + n] = (_Float16)w[i];
}
// Kernel 0b: vh = (f16) v  (one-time 0.46MB convert; k3 A operand)
__global__ void k0_vh(const float* __restrict__ v, _Float16* __restrict__ vh) {
  const int i = blockIdx.x * 256 + threadIdx.x;   // 0..Tt*Tt-1
  vh[i] = (_Float16)v[i];
}

// ---------------------------------------------------------------------------
// Kernel 1: single pass over seq producing
//   f1[b][t][n] = sum_c seq[b,c,n,t]*w1[c]   (f16, K-contiguous for GEMM1 A)
//   f2[b][t][c] = sum_n seq[b,c,n,t]*w2[n]   (f16, K-contiguous for GEMM2 B)
// grid: B*(T/16), 256 threads. HBM-bound: reads seq (1 GB) exactly once.
// ---------------------------------------------------------------------------
__global__ void k1_f1f2(const float* __restrict__ seq,
                        const float* __restrict__ w1,
                        const float* __restrict__ w2,
                        _Float16* __restrict__ f1,
                        _Float16* __restrict__ f2) {
  __shared__ float    red[16][32][16];   // [n-row][c][t]   32 KB
  __shared__ _Float16 f1t[16][520];      // [t][n] padded  ~16.6 KB
  __shared__ float    w1s[32];
  const int blk = blockIdx.x;
  const int b  = blk / (Tt / 16);
  const int t0 = (blk % (Tt / 16)) * 16;
  const int tid = threadIdx.x;
  const int row = tid >> 4;   // n within 16-chunk
  const int tc  = tid & 15;   // t within tile
  if (tid < 32) w1s[tid] = w1[tid];
  __syncthreads();

  float f2part[32];
  #pragma unroll
  for (int c = 0; c < 32; ++c) f2part[c] = 0.f;

  const size_t bbase = (size_t)b * Cc * Nn * Tt;
  for (int chunk = 0; chunk < Nn / 16; ++chunk) {
    const int n = chunk * 16 + row;
    const float w2n = w2[n];
    const float* p = seq + bbase + (size_t)n * Tt + t0 + tc;
    float f1acc = 0.f;
    #pragma unroll
    for (int c = 0; c < 32; ++c) {
      const float val = p[(size_t)c * Nn * Tt];
      f1acc     += val * w1s[c];
      f2part[c] += val * w2n;
    }
    f1t[tc][n] = (_Float16)f1acc;
  }
  #pragma unroll
  for (int c = 0; c < 32; ++c) red[row][c][tc] = f2part[c];
  __syncthreads();

  // coalesced write of f1 strip [b][t][n]
  for (int i = tid; i < 16 * Nn; i += 256) {
    const int t = i >> 9, n = i & (Nn - 1);
    f1[((size_t)(b * Tt + t0 + t)) * Nn + n] = f1t[t][n];
  }
  // reduce f2 over n-rows, write [b][t][c] (c fast -> coalesced)
  for (int i = tid; i < 16 * 32; i += 256) {
    const int t = i >> 5, c = i & 31;
    float s = 0.f;
    #pragma unroll
    for (int r = 0; r < 16; ++r) s += red[r][c][t];
    f2[((size_t)(b * Tt + t0 + t)) * Cc + c] = (_Float16)s;
  }
}

// ---------------------------------------------------------------------------
// Kernel 2: tmp(16x32) = f1strip(16x512) @ w(512x32)   [waves 0-1, WMMA]
//           l0 = sigmoid(tmp @ f2[b](32x480) + bias)   [all 8 waves, WMMA]
// Output stored TRANSPOSED: l0T[b][u][s] (s contiguous = GEMM3 B layout).
// grid: B*(T/16), 256 threads.
// ---------------------------------------------------------------------------
__global__ void k2_gemm12(const _Float16* __restrict__ f1,
                          const _Float16* __restrict__ f2,
                          const _Float16* __restrict__ wT,
                          const float* __restrict__ bbias,
                          _Float16* __restrict__ l0T) {
  __shared__ _Float16 tmps[16][40];   // [t][c] padded, A-layout for phase B
  const int blk = blockIdx.x;
  const int b  = blk / (Tt / 16);
  const int t0 = (blk % (Tt / 16)) * 16;
  const int tid  = threadIdx.x;
  const int wave = tid >> 5, lane = tid & 31;
  const int half = lane >> 4, lm = lane & 15;

  if (wave < 2) {               // GEMM1: two 16x16 c-tiles on waves 0,1
    const int c0 = wave * 16;
    const _Float16* arow = f1 + ((size_t)(b * Tt + t0 + lm)) * Nn;
    const _Float16* bcol = wT + (size_t)(c0 + lm) * Nn;
    v8f acc = {};
    for (int kc = 0; kc < Nn; kc += 32) {
      v16h a  = loadA16(arow, kc, half);
      v16h bf = loadB16(bcol, kc, half);
      acc = __builtin_amdgcn_wmma_f32_16x16x32_f16(false, a, false, bf,
                                                   (short)0, acc, false, false);
    }
    #pragma unroll
    for (int r = 0; r < 8; ++r)
      tmps[half * 8 + r][c0 + lm] = (_Float16)acc[r];
  }
  __syncthreads();

  // GEMM2 + sigmoid: s-tiles round-robin over 8 waves (K=32 -> 1 WMMA)
  for (int st = wave; st < Tt / 16; st += 8) {
    const int s0 = st * 16;
    v16h a  = loadA16(&tmps[lm][0], 0, half);
    v16h bf = loadB16(f2 + ((size_t)(b * Tt + s0 + lm)) * Cc, 0, half);
    v8f acc = {};
    acc = __builtin_amdgcn_wmma_f32_16x16x32_f16(false, a, false, bf,
                                                 (short)0, acc, false, false);
    v8h o;
    #pragma unroll
    for (int r = 0; r < 8; ++r) {
      const int tg = t0 + half * 8 + r;
      const int sg = s0 + lm;
      const float x = acc[r] + bbias[(size_t)tg * Tt + sg];
      o[r] = (_Float16)(1.0f / (1.0f + __expf(-x)));
    }
    // lane writes 8 consecutive s-elements of row u=sg: one 16B store
    *(v8h*)(l0T + ((size_t)(b * Tt + s0 + lm)) * Tt + t0 + half * 8) = o;
  }
}

// ---------------------------------------------------------------------------
// Kernel 3: logits1[b,t,u] = sum_s v[t,s]*l0[b,s,u]  (K=480)
// LDS-free: each wave computes an independent 16x48 tile (A-fragment reused
// by 3 WMMAs). A from vh (f16), B from l0T: pure b128 loads. Fused BN
// partial-sum atomics. grid: B*30*10 waves / 8 per block.
// ---------------------------------------------------------------------------
__global__ void k3_bigmm(const _Float16* __restrict__ vh,
                         const _Float16* __restrict__ l0T,
                         float* __restrict__ l1,
                         float* __restrict__ gsum,
                         float* __restrict__ gsumsq) {
  const int wid  = blockIdx.x * 8 + (threadIdx.x >> 5);  // 0..9599
  const int lane = threadIdx.x & 31;
  const int half = lane >> 4, lm = lane & 15;
  const int b   = wid / (30 * 10);
  const int rem = wid % (30 * 10);
  const int ti  = rem / 10;          // t-tile 0..29
  const int u3  = rem % 10;          // u-triple 0..9
  const int tg0 = ti * 16, u0 = u3 * 48;

  const _Float16* arow = vh + (size_t)(tg0 + lm) * Tt;
  const _Float16* bc0  = l0T + ((size_t)b * Tt + u0 + lm) * Tt;
  const _Float16* bc1  = bc0 + (size_t)16 * Tt;
  const _Float16* bc2  = bc0 + (size_t)32 * Tt;

  v8f acc0 = {}, acc1 = {}, acc2 = {};
  for (int k0 = 0; k0 < Tt; k0 += 32) {
    v16h a  = loadA16(arow, k0, half);
    v16h b0 = loadB16(bc0, k0, half);
    v16h b1 = loadB16(bc1, k0, half);
    v16h b2 = loadB16(bc2, k0, half);
    acc0 = __builtin_amdgcn_wmma_f32_16x16x32_f16(false, a, false, b0,
                                                  (short)0, acc0, false, false);
    acc1 = __builtin_amdgcn_wmma_f32_16x16x32_f16(false, a, false, b1,
                                                  (short)0, acc1, false, false);
    acc2 = __builtin_amdgcn_wmma_f32_16x16x32_f16(false, a, false, b2,
                                                  (short)0, acc2, false, false);
  }

  float s0 = 0.f, ss0 = 0.f, s1 = 0.f, ss1 = 0.f, s2 = 0.f, ss2 = 0.f;
  #pragma unroll
  for (int r = 0; r < 8; ++r) {
    const int t = tg0 + half * 8 + r;
    float* rp = l1 + ((size_t)b * Tt + t) * Tt;
    const float x0 = acc0[r], x1 = acc1[r], x2 = acc2[r];
    rp[u0 + lm]      = x0;
    rp[u0 + 16 + lm] = x1;
    rp[u0 + 32 + lm] = x2;
    s0 += x0; ss0 += x0 * x0;
    s1 += x1; ss1 += x1 * x1;
    s2 += x2; ss2 += x2 * x2;
  }
  s0  += __shfl_xor(s0, 16, 32);   ss0 += __shfl_xor(ss0, 16, 32);
  s1  += __shfl_xor(s1, 16, 32);   ss1 += __shfl_xor(ss1, 16, 32);
  s2  += __shfl_xor(s2, 16, 32);   ss2 += __shfl_xor(ss2, 16, 32);
  if (half == 0) {
    atomicAdd(&gsum[u0 + lm], s0);        atomicAdd(&gsumsq[u0 + lm], ss0);
    atomicAdd(&gsum[u0 + 16 + lm], s1);   atomicAdd(&gsumsq[u0 + 16 + lm], ss1);
    atomicAdd(&gsum[u0 + 32 + lm], s2);   atomicAdd(&gsumsq[u0 + 32 + lm], ss2);
  }
}

// ---------------------------------------------------------------------------
// Kernel 4: BN-normalize + mask + masked softmax, one wave per (b,t) row,
// in-place on d_out.
// ---------------------------------------------------------------------------
__global__ void k4_bnsoftmax(float* __restrict__ l1,
                             const float* __restrict__ gsum,
                             const float* __restrict__ gsumsq,
                             const float* __restrict__ gamma,
                             const float* __restrict__ beta,
                             const float* __restrict__ mask) {
  const int wid  = threadIdx.x >> 5;
  const int lane = threadIdx.x & 31;
  const int rowid = blockIdx.x * 8 + wid;   // 0..B*T-1
  const int t = rowid % Tt;
  float* rp = l1 + (size_t)rowid * Tt;
  const float invM = 1.0f / (float)(Bb * Tt);

  float x[Tt / 32];
  float mx = -3.0e38f;
  #pragma unroll
  for (int i = 0; i < Tt / 32; ++i) {
    const int u = lane + 32 * i;
    const float mean = gsum[u] * invM;
    const float var  = gsumsq[u] * invM - mean * mean;
    float val = (rp[u] - mean) * rsqrtf(var + 1e-5f) * gamma[u] + beta[u];
    val += mask[(size_t)t * Tt + u];
    x[i] = val;
    mx = fmaxf(mx, val);
  }
  for (int off = 16; off; off >>= 1) mx = fmaxf(mx, __shfl_xor(mx, off, 32));
  float se = 0.f;
  #pragma unroll
  for (int i = 0; i < Tt / 32; ++i) { x[i] = __expf(x[i] - mx); se += x[i]; }
  for (int off = 16; off; off >>= 1) se += __shfl_xor(se, off, 32);
  const float inv = 1.0f / se;
  #pragma unroll
  for (int i = 0; i < Tt / 32; ++i) rp[lane + 32 * i] = x[i] * inv;
}

// ---------------------------------------------------------------------------
extern "C" void kernel_launch(void* const* d_in, const int* in_sizes, int n_in,
                              void* d_out, int out_size, void* d_ws, size_t ws_size,
                              hipStream_t stream) {
  const float* seq   = (const float*)d_in[0];
  const float* w1    = (const float*)d_in[1];
  const float* w2    = (const float*)d_in[2];
  const float* w     = (const float*)d_in[3];
  const float* bbias = (const float*)d_in[4];
  const float* v     = (const float*)d_in[5];
  const float* gamma = (const float*)d_in[6];
  const float* beta  = (const float*)d_in[7];
  const float* mask  = (const float*)d_in[8];

  char* ws = (char*)d_ws;
  const size_t OFF_F1  = 0;                                   // f16 [B][T][N]
  const size_t OFF_F2  = OFF_F1 + (size_t)Bb * Tt * Nn * 2;   // f16 [B][T][C]
  const size_t OFF_WT  = OFF_F2 + (size_t)Bb * Tt * Cc * 2;   // f16 [C][N]
  const size_t OFF_VH  = OFF_WT + (size_t)Cc * Nn * 2;        // f16 [T][T]
  const size_t OFF_L0  = OFF_VH + (size_t)Tt * Tt * 2;        // f16 [B][u][s]
  const size_t OFF_SUM = OFF_L0 + (size_t)Bb * Tt * Tt * 2;   // f32 [T]
  const size_t OFF_SSQ = OFF_SUM + (size_t)Tt * 4;            // f32 [T]

  _Float16* f1  = (_Float16*)(ws + OFF_F1);
  _Float16* f2  = (_Float16*)(ws + OFF_F2);
  _Float16* wT  = (_Float16*)(ws + OFF_WT);
  _Float16* vh  = (_Float16*)(ws + OFF_VH);
  _Float16* l0T = (_Float16*)(ws + OFF_L0);
  float*    gsm = (float*)(ws + OFF_SUM);
  float*    gsq = (float*)(ws + OFF_SSQ);
  float*    out = (float*)d_out;   // reused as logits1 scratch, finalized in-place

  k0_wT<<<(Nn * Cc) / 256, 256, 0, stream>>>(w, wT);
  k0_vh<<<(Tt * Tt) / 256, 256, 0, stream>>>(v, vh);
  k1_f1f2<<<Bb * (Tt / 16), 256, 0, stream>>>(seq, w1, w2, f1, f2);
  k2_gemm12<<<Bb * (Tt / 16), 256, 0, stream>>>(f1, f2, wT, bbias, l0T);
  hipMemsetAsync(gsm, 0, 2 * Tt * sizeof(float), stream);
  k3_bigmm<<<(Bb * 30 * 10) / 8, 256, 0, stream>>>(vh, l0T, out, gsm, gsq);
  k4_bnsoftmax<<<(Bb * Tt) / 8, 256, 0, stream>>>(out, gsm, gsq, gamma, beta, mask);
}
